// HeadWordProbEncoder_7430293422145
// MI455X (gfx1250) — matrix operand order repaired and checked by reading, not compiled, not run on hardware
//
#include <hip/hip_runtime.h>
#include <hip/hip_bf16.h>

// ---------------------------------------------------------------------------
// HeadWordProbEncoder on MI455X (gfx1250).
// All heavy contractions run as bf16 WMMA (v_wmma_f32_16x16x32_bf16, f32 acc).
// DAMP=0 / STEP=1 / REG=1 collapse all cache terms; mask==1 everywhere.
// Each wave computes a 16x64 output tile with a double-buffered K loop:
// loads for K-step n+1 are issued before the 4 WMMAs of step n, so
// s_wait_loadcnt covers a full iteration of latency instead of draining.
// ---------------------------------------------------------------------------

typedef __attribute__((ext_vector_type(16))) __bf16 v16bf;
typedef __attribute__((ext_vector_type(8)))  float  v8f;

#define B_   4
#define L_   128
#define D_   128
#define H_   8
#define V_   32000
#define NEGC 1e9f

// ---------------------------------------------------------------------------
// Generic batched bf16 GEMM: one wave computes a 16x64 f32 tile, K step 32.
//   A: bf16 row-major, element (m,k) at  offA + m*lda + k   (k contiguous)
//   B: bf16 row-major, element (k,n) at  offB + k*ldb + n   (n contiguous)
//   C: f32 and/or bf16, element (m,n) at offC + m*ldc + n*csc
// Batch offsets: y = blockIdx.y decomposed as (y/dY, y%dY), plus blockIdx.z:
//   off* = (y/dY)*s0 + (y%dY)*s1 + z*sz   per operand.
// Requires M % 16 == 0, N % 64 == 0, K % 32 == 0.
// ---------------------------------------------------------------------------
__global__ __launch_bounds__(256) void gemm_bf16_wmma(
    const __bf16* __restrict__ A, const __bf16* __restrict__ B,
    float* __restrict__ Cf, __bf16* __restrict__ Ch,
    int M, int N, int K, int lda, int ldb, int ldc, int csc, int dY,
    long a0, long a1, long az,
    long b0, long b1, long bz,
    long c0, long c1, long cz)
{
    const int wavesPerBlock = blockDim.x >> 5;
    const int wid = blockIdx.x * wavesPerBlock + (threadIdx.x >> 5);
    const int tilesN = N >> 6;                   // 64-wide N tiles
    const int tilesM = M >> 4;
    if (wid >= tilesM * tilesN) return;          // whole-wave uniform exit
    const int tm = wid / tilesN;
    const int tn = wid % tilesN;

    const int y  = blockIdx.y, zb = blockIdx.z;
    const long offA = (long)(y / dY) * a0 + (long)(y % dY) * a1 + (long)zb * az;
    const long offB = (long)(y / dY) * b0 + (long)(y % dY) * b1 + (long)zb * bz;
    const long offC = (long)(y / dY) * c0 + (long)(y % dY) * c1 + (long)zb * cz;

    const int lane = threadIdx.x & 31;
    const int mrow = lane & 15;        // A: lane -> M row (both half-waves)
    const int hi   = lane >> 4;        // half-wave select for K packing

    // Strength-reduced pointers: one multiply each, then pure increments.
    // A fragment: elems 0..7 = K[k0+hi*8 .. +7], elems 8..15 = K[k0+16+hi*8 .. +7]
    const __bf16* ap = A + offA + (long)(tm * 16 + mrow) * lda + hi * 8;
    // B fragments: lane&15 = N column; elems 0..7 = K[k0+hi*8 .. +7], elems 8..15 = K[k0+16+hi*8 .. +7]
    const __bf16* bp = B + offB + tn * 64 + (lane & 15) + (long)(hi * 8) * ldb;
    const long bstep = 32L * ldb;

    v8f acc[4] = {{}, {}, {}, {}};
    v16bf a, bfrag[4];

    // Prologue: loads for K-step 0.
    ((int4*)&a)[0] = *(const int4*)(ap);
    ((int4*)&a)[1] = *(const int4*)(ap + 16);
#pragma unroll
    for (int t = 0; t < 4; ++t) {
        for (int s = 0; s < 8; ++s) {
            bfrag[t][s]     = bp[(long)s * ldb + t * 16];
            bfrag[t][8 + s] = bp[(long)(16 + s) * ldb + t * 16];
        }
    }

    // Steady state: prefetch step n+1, then 4 WMMAs on step n.
    for (int k0 = 32; k0 < K; k0 += 32) {
        ap += 32;
        bp += bstep;
        v16bf a2, b2[4];
        ((int4*)&a2)[0] = *(const int4*)(ap);
        ((int4*)&a2)[1] = *(const int4*)(ap + 16);
#pragma unroll
        for (int t = 0; t < 4; ++t) {
            for (int s = 0; s < 8; ++s) {
                b2[t][s]     = bp[(long)s * ldb + t * 16];
                b2[t][8 + s] = bp[(long)(16 + s) * ldb + t * 16];
            }
        }
#pragma unroll
        for (int t = 0; t < 4; ++t)
            acc[t] = __builtin_amdgcn_wmma_f32_16x16x32_bf16(
                false, a, false, bfrag[t], (short)0, acc[t], false, false);
        a = a2;
#pragma unroll
        for (int t = 0; t < 4; ++t) bfrag[t] = b2[t];
    }
    // Epilogue: last K-step.
#pragma unroll
    for (int t = 0; t < 4; ++t)
        acc[t] = __builtin_amdgcn_wmma_f32_16x16x32_bf16(
            false, a, false, bfrag[t], (short)0, acc[t], false, false);

    // C layout: vgpr r, lanes 0-15 -> M=r, lanes 16-31 -> M=8+r; N = lane&15
    const int rbase = tm * 16 + hi * 8;
#pragma unroll
    for (int t = 0; t < 4; ++t) {
        const int col = tn * 64 + t * 16 + (lane & 15);
#pragma unroll
        for (int r = 0; r < 8; ++r) {
            const long off = offC + (long)(rbase + r) * ldc + (long)col * csc;
            if (Cf) Cf[off] = acc[t][r];
            if (Ch) Ch[off] = (__bf16)acc[t][r];
        }
    }
}

// ---------------------------------------------------------------------------
// One-time prep kernels
// ---------------------------------------------------------------------------
__global__ void prep_embed_kernel(const float* __restrict__ E,
                                  __bf16* __restrict__ Eb,
                                  __bf16* __restrict__ ETb)
{
    const long n = (long)V_ * D_;
    for (long idx = (long)blockIdx.x * blockDim.x + threadIdx.x; idx < n;
         idx += (long)gridDim.x * blockDim.x) {
        const float e = E[idx];
        const int  d = (int)(idx % D_);
        const long v = idx / D_;
        Eb[idx] = (__bf16)e;
        ETb[(long)d * V_ + v] = (__bf16)e;   // E^T for q_z @ E^T
    }
}

__global__ __launch_bounds__(128) void colmean_kernel(const float* __restrict__ E,
                                                      float* __restrict__ pre)
{
    const int d = threadIdx.x;
    float s = 0.f;
    for (int v = 0; v < V_; ++v) s += E[(long)v * D_ + d];
    pre[d] = s / (float)V_;                  // softmax(ones) @ E == colmean(E)
}

// ternary[k][a][b][c] -> TA[k][a][c][b] (b fast) and TB[k][b][c][a] (a fast)
__global__ void perm_ternary_kernel(const float* __restrict__ T,
                                    __bf16* __restrict__ TA,
                                    __bf16* __restrict__ TB)
{
    const int n = 2 * D_ * D_ * H_;
    for (int idx = blockIdx.x * blockDim.x + threadIdx.x; idx < n;
         idx += gridDim.x * blockDim.x) {
        const int c = idx % H_;
        const int b = (idx / H_) % D_;
        const int a = (idx / (H_ * D_)) % D_;
        const int k = idx / (H_ * D_ * D_);
        const __bf16 v = (__bf16)T[idx];
        TA[(((long)(k * D_ + a) * H_ + c) * D_) + b] = v;
        TB[(((long)(k * D_ + b) * H_ + c) * D_) + a] = v;
    }
}

// ---------------------------------------------------------------------------
// Row softmax kernels (block = one row, LDS tree reductions)
// ---------------------------------------------------------------------------
__global__ __launch_bounds__(128) void init_qz_kernel(
    const float* __restrict__ x, const int* __restrict__ m_mask,
    const float* __restrict__ pre,
    __bf16* __restrict__ Qz, __bf16* __restrict__ QzT)
{
    const int row = blockIdx.x;              // z*L + i
    const int z   = row / L_;
    const int d   = threadIdx.x;
    const float mm = (float)m_mask[row];
    const float v  = x[(long)row * D_ + d] * (1.f - mm) + pre[d] * mm;
    __shared__ float sm[D_];
    sm[d] = v; __syncthreads();
    for (int s = 64; s > 0; s >>= 1) { if (d < s) sm[d] = fmaxf(sm[d], sm[d + s]); __syncthreads(); }
    const float mx = sm[0]; __syncthreads();
    const float e = __expf(v - mx);
    sm[d] = e; __syncthreads();
    for (int s = 64; s > 0; s >>= 1) { if (d < s) sm[d] += sm[d + s]; __syncthreads(); }
    const float p = e / sm[0];
    Qz[(long)row * D_ + d] = (__bf16)p;
    QzT[(long)z * D_ * L_ + (long)d * L_ + (row % L_)] = (__bf16)p;
}

// q_h row softmax with distance masks fused; emits masked m and m^T in bf16.
// Ff layout: [k][z*H+c][i][j]
__global__ __launch_bounds__(128) void h_softmax_kernel(
    const float* __restrict__ Ff, __bf16* __restrict__ Mb, __bf16* __restrict__ MTb)
{
    const int bi = blockIdx.x;               // (z*H+c)*L + i
    const int i  = bi % L_;
    const int zc = bi / L_;
    const int j  = threadIdx.x;
    const long base0  = (long)zc * (L_ * L_) + (long)i * L_ + j;
    const long strK   = (long)B_ * H_ * L_ * L_;
    const float f0 = Ff[base0];
    const float f1 = Ff[base0 + strK];
    const float v = (j > i) ? f0 : ((j < i) ? f1 : -NEGC);   // q_h with -eye*NEG
    __shared__ float sm[L_];
    sm[j] = v; __syncthreads();
    for (int s = 64; s > 0; s >>= 1) { if (j < s) sm[j] = fmaxf(sm[j], sm[j + s]); __syncthreads(); }
    const float mx = sm[0]; __syncthreads();
    const float e = __expf(v - mx);
    sm[j] = e; __syncthreads();
    for (int s = 64; s > 0; s >>= 1) { if (j < s) sm[j] += sm[j + s]; __syncthreads(); }
    const float p = e / sm[0];
    const float m0 = (j > i) ? p : 0.f;      // m = q_hn * distmask[k]
    const float m1 = (j < i) ? p : 0.f;
    Mb[base0]        = (__bf16)m0;
    Mb[base0 + strK] = (__bf16)m1;
    const long baseT = (long)zc * (L_ * L_) + (long)j * L_ + i;
    MTb[baseT]        = (__bf16)m0;
    MTb[baseT + strK] = (__bf16)m1;
}

__global__ __launch_bounds__(256) void softmax_v_kernel(
    const float* __restrict__ qx, __bf16* __restrict__ px)
{
    const int row = blockIdx.x;              // z*L+i
    const float* r = qx + (long)row * V_;
    __bf16* o = px + (long)row * V_;
    const int t = threadIdx.x;
    float mx = -3.4e38f;
    for (int v = t; v < V_; v += 256) mx = fmaxf(mx, r[v]);
    __shared__ float sm[256];
    sm[t] = mx; __syncthreads();
    for (int s = 128; s > 0; s >>= 1) { if (t < s) sm[t] = fmaxf(sm[t], sm[t + s]); __syncthreads(); }
    mx = sm[0]; __syncthreads();
    float sum = 0.f;
    for (int v = t; v < V_; v += 256) sum += __expf(r[v] - mx);
    sm[t] = sum; __syncthreads();
    for (int s = 128; s > 0; s >>= 1) { if (t < s) sm[t] += sm[t + s]; __syncthreads(); }
    const float inv = 1.f / sm[0];
    for (int v = t; v < V_; v += 256) o[v] = (__bf16)(__expf(r[v] - mx) * inv);
}

__global__ void fill_xmsg_kernel(const float* __restrict__ pre, float* __restrict__ xmsg)
{
    const int idx = blockIdx.x * blockDim.x + threadIdx.x;
    if (idx < B_ * L_ * D_) xmsg[idx] = pre[idx % D_];   // iter-0 X message
}

__global__ __launch_bounds__(128) void z_update_kernel(
    const float* __restrict__ x, const int* __restrict__ m_mask,
    const float* __restrict__ xmsg, const float* __restrict__ G1,
    const float* __restrict__ G2,
    __bf16* __restrict__ Qz, __bf16* __restrict__ QzT)
{
    const int row = blockIdx.x;              // z*L+i
    const int z   = row / L_;
    const int d   = threadIdx.x;
    const float mm = (float)m_mask[row];
    const long idx = (long)row * D_ + d;
    const float v = x[idx] * (1.f - mm) + xmsg[idx] * mm + G1[idx] + G2[idx];
    __shared__ float sm[D_];
    sm[d] = v; __syncthreads();
    for (int s = 64; s > 0; s >>= 1) { if (d < s) sm[d] = fmaxf(sm[d], sm[d + s]); __syncthreads(); }
    const float mx = sm[0]; __syncthreads();
    const float e = __expf(v - mx);
    sm[d] = e; __syncthreads();
    for (int s = 64; s > 0; s >>= 1) { if (d < s) sm[d] += sm[d + s]; __syncthreads(); }
    const float p = e / sm[0];
    Qz[idx] = (__bf16)p;
    QzT[(long)z * D_ * L_ + (long)d * L_ + (row % L_)] = (__bf16)p;
}

// ---------------------------------------------------------------------------
extern "C" void kernel_launch(void* const* d_in, const int* in_sizes, int n_in,
                              void* d_out, int out_size, void* d_ws, size_t ws_size,
                              hipStream_t stream)
{
    (void)in_sizes; (void)n_in; (void)out_size; (void)ws_size;
    const float* x      = (const float*)d_in[0];
    /* d_in[1] = mask: all ones -> keep/inv2d are no-ops */
    const float* E      = (const float*)d_in[2];
    const int*   m_mask = (const int*)d_in[3];
    const float* Tern   = (const float*)d_in[4];
    float* qx = (float*)d_out;               // q_x [B,L,V]

    char* w = (char*)d_ws;
    auto alloc = [&](size_t bytes) -> char* {
        char* p = w; w += (bytes + 255) & ~(size_t)255; return p;
    };
    __bf16* Eb   = (__bf16*)alloc((size_t)V_ * D_ * 2);        // E  bf16 [V,D]
    __bf16* ETb  = (__bf16*)alloc((size_t)V_ * D_ * 2);        // E^T bf16 [D,V]
    __bf16* TA   = (__bf16*)alloc((size_t)2 * D_ * H_ * D_ * 2); // [k][a][c][b]
    __bf16* TB   = (__bf16*)alloc((size_t)2 * D_ * H_ * D_ * 2); // [k][b][c][a]
    float*  pre  = (float*) alloc(D_ * 4);
    __bf16* Qz   = (__bf16*)alloc((size_t)B_ * L_ * D_ * 2);   // [(z,i)][d]
    __bf16* QzT  = (__bf16*)alloc((size_t)B_ * D_ * L_ * 2);   // [z][d][i]
    __bf16* t1p  = (__bf16*)alloc((size_t)2 * 512 * 1024 * 2); // [k][(z,i)][(c,b)]
    float*  Ff   = (float*) alloc((size_t)2 * B_ * H_ * L_ * L_ * 4);
    __bf16* Mb   = (__bf16*)alloc((size_t)2 * B_ * H_ * L_ * L_ * 2);
    __bf16* MTb  = (__bf16*)alloc((size_t)2 * B_ * H_ * L_ * L_ * 2);
    __bf16* S1   = (__bf16*)alloc((size_t)512 * 2048 * 2);     // [(z,i)][(k,b,c)]
    __bf16* S2   = (__bf16*)alloc((size_t)512 * 2048 * 2);     // [(z,j)][(k,a,c)]
    float*  G1b  = (float*) alloc((size_t)512 * 128 * 4);
    float*  G2b  = (float*) alloc((size_t)512 * 128 * 4);
    float*  xmsg = (float*) alloc((size_t)512 * 128 * 4);
    __bf16* px   = (__bf16*)alloc((size_t)512 * (size_t)V_ * 2);

    // one-time prep
    prep_embed_kernel<<<2048, 256, 0, stream>>>(E, Eb, ETb);
    colmean_kernel<<<1, 128, 0, stream>>>(E, pre);
    perm_ternary_kernel<<<512, 256, 0, stream>>>(Tern, TA, TB);
    init_qz_kernel<<<B_ * L_, 128, 0, stream>>>(x, m_mask, pre, Qz, QzT);

    for (int it = 0; it < 4; ++it) {
        // t1p[k] = Qz [512x128] @ TA[k] [128x1024]  (bf16 out)
        // tiles: 32 x 16 = 512 waves -> 64 blocks
        gemm_bf16_wmma<<<dim3(64, 2, 1), 256, 0, stream>>>(
            Qz, TA, nullptr, t1p, 512, 1024, 128, 128, 1024, 1024, 1, 1,
            0, 0, 0,  (long)D_ * H_ * D_, 0, 0,  (long)512 * 1024, 0, 0);

        // F[k][z][c] = t1p_slice [128x128] @ QzT[z] [128x128]  (f32 out)
        // tiles/batch: 8 x 2 = 16 waves -> 2 blocks
        gemm_bf16_wmma<<<dim3(2, 32, 2), 256, 0, stream>>>(
            t1p, QzT, Ff, nullptr, 128, 128, 128, 1024, 128, 128, 1, /*dY=*/8,
            /*A z,c,k*/ 131072, 128, 524288,
            /*B*/       16384, 0, 0,
            /*C*/       131072, 16384, 524288);

        h_softmax_kernel<<<B_ * H_ * L_, 128, 0, stream>>>(Ff, Mb, MTb);

        // S1[(z,i)][(k,b,c)] = Mb[k][z][c] @ Qz[z]   (interleaved bf16 out, csc=8)
        gemm_bf16_wmma<<<dim3(2, 8, 8), 256, 0, stream>>>(
            Mb, Qz, nullptr, S1, 128, 128, 128, 128, 128, 2048, 8, /*dY=*/2,
            /*A z,k,c*/ 131072, 524288, 16384,
            /*B*/       16384, 0, 0,
            /*C*/       262144, 1024, 1);
        // S2[(z,j)][(k,a,c)] = MTb[k][z][c] @ Qz[z]
        gemm_bf16_wmma<<<dim3(2, 8, 8), 256, 0, stream>>>(
            MTb, Qz, nullptr, S2, 128, 128, 128, 128, 128, 2048, 8, /*dY=*/2,
            131072, 524288, 16384,  16384, 0, 0,  262144, 1024, 1);

        // G1 = S1 [512x2048] @ TB [2048x128];  G2 = S2 @ TA
        // tiles: 32 x 2 = 64 waves -> 8 blocks
        gemm_bf16_wmma<<<dim3(8, 1, 1), 256, 0, stream>>>(
            S1, TB, G1b, nullptr, 512, 128, 2048, 2048, 128, 128, 1, 1,
            0, 0, 0, 0, 0, 0, 0, 0, 0);
        gemm_bf16_wmma<<<dim3(8, 1, 1), 256, 0, stream>>>(
            S2, TA, G2b, nullptr, 512, 128, 2048, 2048, 128, 128, 1, 1,
            0, 0, 0, 0, 0, 0, 0, 0, 0);

        // X message into Z: softmax(q_x) @ E  (iter 0: uniform -> colmean)
        if (it == 0) {
            fill_xmsg_kernel<<<256, 256, 0, stream>>>(pre, xmsg);
        } else {
            softmax_v_kernel<<<B_ * L_, 256, 0, stream>>>(qx, px);
            gemm_bf16_wmma<<<dim3(8, 1, 1), 256, 0, stream>>>(
                px, Eb, xmsg, nullptr, 512, 128, V_, V_, 128, 128, 1, 1,
                0, 0, 0, 0, 0, 0, 0, 0, 0);
        }

        z_update_kernel<<<B_ * L_, 128, 0, stream>>>(x, m_mask, xmsg, G1b, G2b, Qz, QzT);

        // q_x = Qz [512x128] @ E^T [128x32000]  -> d_out (f32)
        // tiles: 32 x 500 = 16000 waves -> 2000 blocks
        gemm_bf16_wmma<<<dim3(2000, 1, 1), 256, 0, stream>>>(
            Qz, ETb, qx, nullptr, 512, V_, 128, 128, V_, V_, 1, 1,
            0, 0, 0, 0, 0, 0, 0, 0, 0);
    }
}
